// FunnelAeDecoder_58110907514926
// MI455X (gfx1250) — matrix operand order repaired and checked by reading, not compiled, not run on hardware
//
#include <hip/hip_runtime.h>
#include <math.h>

// ---------------------------------------------------------------------------
// CDNA5 (gfx1250) wave32 WMMA types
// ---------------------------------------------------------------------------
typedef __attribute__((ext_vector_type(16))) __bf16 v16bf;
typedef __attribute__((ext_vector_type(8)))  __bf16 v8bf;
typedef __attribute__((ext_vector_type(8)))  float  v8f;

__device__ inline v8f wmma_bf16(v16bf a, v16bf b, v8f c) {
  return __builtin_amdgcn_wmma_f32_16x16x32_bf16(false, a, false, b, (short)0, c,
                                                 false, false);
}

#define D_MODEL 1024
#define NHEAD 16
#define DHEAD 64
#define BATCH 2
#define ATT_SCALE 0.125f
#define INF_PEN 1000000.0f

// ===========================================================================
// GEMM: C[M,1024] = scale * (A[M,1024] @ W[1024,1024] + bias)
// 64x64 block tile, staged K-slab of 64 (2 x K=32 WMMA chunks per barrier),
// 4 waves (128 threads), bf16 WMMA f32-acc. M is always a multiple of 64
// in this model, so there are NO bounds guards (no exec-mask divergence).
// ===========================================================================
#define LDS_STRIDE 72   // 64 k + 8 pad (bf16) -> 144B rows, 16B aligned chunks

__global__ __launch_bounds__(128)
void gemm_kernel(const float* __restrict__ A, const float* __restrict__ W,
                 const float* __restrict__ bias, float* __restrict__ C,
                 int M, float scale)
{
  __shared__ __align__(16) __bf16 As[64 * LDS_STRIDE];
  __shared__ __align__(16) __bf16 Wt[64 * LDS_STRIDE]; // transposed: [n][k]

  const int tid  = threadIdx.x;
  const int lane = tid & 31;
  const int wave = tid >> 5;
  const int wr = wave >> 1, wc = wave & 1;
  const int ln = lane & 15, half = lane >> 4;
  const int rowBase = blockIdx.y * 64;
  const int colBase = blockIdx.x * 64;

  v8f acc[2][2];
  acc[0][0] = (v8f){}; acc[0][1] = (v8f){};
  acc[1][0] = (v8f){}; acc[1][1] = (v8f){};

  for (int k0 = 0; k0 < 1024; k0 += 64) {
    // speculative prefetch of next K-slab (global_prefetch_b8)
    if (k0 + 64 < 1024) {
      int pr = tid & 63;
      __builtin_prefetch(A + (size_t)(rowBase + pr) * 1024 + k0 + 64, 0, 0);
      __builtin_prefetch(W + (size_t)(k0 + 64 + pr) * 1024 + colBase, 0, 0);
    }
    __syncthreads();
    // A tile: 64 rows x 64 k (f32 -> bf16). 1024 float4 / 128 threads = 8.
    for (int it = 0; it < 8; ++it) {
      int q = tid + it * 128;
      int r = q >> 4, c4 = (q & 15) << 2;
      float4 v = *(const float4*)(A + (size_t)(rowBase + r) * 1024 + k0 + c4);
      __bf16* dst = &As[r * LDS_STRIDE + c4];
      dst[0] = (__bf16)v.x; dst[1] = (__bf16)v.y;
      dst[2] = (__bf16)v.z; dst[3] = (__bf16)v.w;
    }
    // W tile: 64 k x 64 n, stored transposed Wt[n][k].
    for (int it = 0; it < 8; ++it) {
      int q = tid + it * 128;
      int r = q >> 4, c4 = (q & 15) << 2;
      float4 v = *(const float4*)(W + (size_t)(k0 + r) * 1024 + colBase + c4);
      Wt[(c4 + 0) * LDS_STRIDE + r] = (__bf16)v.x;
      Wt[(c4 + 1) * LDS_STRIDE + r] = (__bf16)v.y;
      Wt[(c4 + 2) * LDS_STRIDE + r] = (__bf16)v.z;
      Wt[(c4 + 3) * LDS_STRIDE + r] = (__bf16)v.w;
    }
    __syncthreads();

    for (int kk = 0; kk < 2; ++kk) {
      const int ko = kk * 32;
      v16bf afr[2], bfr[2];
      for (int mi = 0; mi < 2; ++mi) {
        int r = wr * 32 + mi * 16 + ln;
        v8bf lo = *(const v8bf*)&As[r * LDS_STRIDE + ko + half * 8];
        v8bf hi = *(const v8bf*)&As[r * LDS_STRIDE + ko + 16 + half * 8];
        afr[mi] = __builtin_shufflevector(lo, hi,
                    0,1,2,3,4,5,6,7,8,9,10,11,12,13,14,15);
      }
      for (int nj = 0; nj < 2; ++nj) {
        int cc = wc * 32 + nj * 16 + ln;
        v8bf lo = *(const v8bf*)&Wt[cc * LDS_STRIDE + ko + half * 8];
        v8bf hi = *(const v8bf*)&Wt[cc * LDS_STRIDE + ko + 16 + half * 8];
        bfr[nj] = __builtin_shufflevector(lo, hi,
                    0,1,2,3,4,5,6,7,8,9,10,11,12,13,14,15);
      }
      for (int mi = 0; mi < 2; ++mi)
        for (int nj = 0; nj < 2; ++nj)
          acc[mi][nj] = wmma_bf16(afr[mi], bfr[nj], acc[mi][nj]);
    }
  }

  for (int nj = 0; nj < 2; ++nj) {
    int col = colBase + wc * 32 + nj * 16 + ln;
    float bv = bias ? bias[col] : 0.f;
    for (int mi = 0; mi < 2; ++mi) {
      for (int e = 0; e < 8; ++e) {
        int grow = rowBase + wr * 32 + mi * 16 + e + 8 * half;
        C[(size_t)grow * 1024 + col] = scale * (acc[mi][nj][e] + bv);
      }
    }
  }
}

// ===========================================================================
// Trig table: Trig[t, d] for t in [0, 2*Lq): delta = sq*(t - Lq)
//   d < 512 : sinsign * sin(delta * w_d),  else cos(delta * w_{d-512})
// ===========================================================================
__global__ void trig_kernel(float* __restrict__ T, int Lq, int sq, float sinsign)
{
  int d = blockIdx.x * 256 + threadIdx.x;
  int t = blockIdx.y;
  float delta = (float)(sq * (t - Lq));
  int dd = (d < 512) ? d : d - 512;
  float w = __expf(-(float)dd * (9.2103403720f / 512.0f)); // ln(10000)/512
  float a = delta * w;
  T[(size_t)t * 1024 + d] = (d < 512) ? sinsign * __sinf(a) : __cosf(a);
}

// ===========================================================================
// Upsample: out[2t]=h[t], out[2t+1]=0.5*(h[t]+h[t-1 (wrap)])
// ===========================================================================
__global__ void upsample_kernel(const float* __restrict__ h, float* __restrict__ out,
                                int Lin)
{
  size_t idx = (size_t)blockIdx.x * blockDim.x + threadIdx.x;
  size_t total = (size_t)BATCH * Lin * D_MODEL;
  if (idx >= total) return;
  int d = idx % D_MODEL;
  int t = (idx / D_MODEL) % Lin;
  int b = idx / ((size_t)D_MODEL * Lin);
  float cur  = h[((size_t)b * Lin + t) * D_MODEL + d];
  int tm = (t == 0) ? (Lin - 1) : (t - 1);
  float prev = h[((size_t)b * Lin + tm) * D_MODEL + d];
  size_t ob = (((size_t)b * (2 * Lin)) + 2 * t) * D_MODEL + d;
  out[ob] = cur;
  out[ob + D_MODEL] = 0.5f * (cur + prev);
}

// ===========================================================================
// LayerNorm(residual + x) * g + b  over rows of 1024
// ===========================================================================
__global__ __launch_bounds__(256)
void ln_kernel(const float* __restrict__ x, const float* __restrict__ res,
               const float* __restrict__ g, const float* __restrict__ bta,
               float* __restrict__ out)
{
  int row = blockIdx.x;
  int tid = threadIdx.x;
  const float* xr = x + (size_t)row * 1024;
  const float* rr = res + (size_t)row * 1024;
  float vals[4];
  float s = 0.f, ss = 0.f;
  for (int i = 0; i < 4; ++i) {
    int c = tid + i * 256;
    float y = xr[c] + rr[c];
    vals[i] = y; s += y; ss += y * y;
  }
  for (int off = 1; off < 32; off <<= 1) {
    s  += __shfl_xor(s, off);
    ss += __shfl_xor(ss, off);
  }
  __shared__ float sh1[8], sh2[8];
  if ((tid & 31) == 0) { sh1[tid >> 5] = s; sh2[tid >> 5] = ss; }
  __syncthreads();
  s = 0.f; ss = 0.f;
  for (int i = 0; i < 8; ++i) { s += sh1[i]; ss += sh2[i]; }
  float mu  = s * (1.f / 1024.f);
  float var = ss * (1.f / 1024.f) - mu * mu;
  float rstd = rsqrtf(var + 1e-9f);
  for (int i = 0; i < 4; ++i) {
    int c = tid + i * 256;
    out[(size_t)row * 1024 + c] = (vals[i] - mu) * rstd * g[c] + bta[c];
  }
}

// ===========================================================================
// Fused relative-position attention, flash-style, one wave per 16-query tile.
//   score = qw.kT + P-gather(rel table) + token-type + mask ; softmax ; @V
// ===========================================================================
__global__ __launch_bounds__(32)
void attn_kernel(const float* __restrict__ q, const float* __restrict__ k,
                 const float* __restrict__ v, const float* __restrict__ Rtab,
                 const float* __restrict__ rw, const float* __restrict__ rr,
                 const float* __restrict__ rs, const float* __restrict__ seg,
                 const int* __restrict__ ids, const float* __restrict__ amask,
                 float* __restrict__ outv,
                 int Lq, int Lk, int sq, int sk, int cj)
{
  const int itile = blockIdx.x;
  const int n     = blockIdx.y;
  const int b     = blockIdx.z;
  const int lane  = threadIdx.x;
  const int ln    = lane & 15;
  const int half  = lane >> 4;

  __shared__ float P_lds[16 * 48];
  __shared__ float p_lds[16 * 16];
  __shared__ float tt_lds[2][16];
  __shared__ float mr_lds[16];
  __shared__ int   idq_lds[16];

  const int NPB = (cj == 1) ? 2 : 3;

  // ---- per-block precompute: A fragments for content (qw) and pos (qr) ----
  const int arow = itile * 16 + ln;                       // A-matrix row = ln
  const float* qrow = q + ((size_t)(b * Lq + arow)) * 1024 + n * 64;
  v16bf a_c[2], a_r[2];
  for (int kk = 0; kk < 2; ++kk) {
    v16bf ac, ar;
    for (int j = 0; j < 8; ++j) {
      int h = kk * 32 + half * 16 + j;
      float qv = qrow[h];
      ac[j] = (__bf16)(qv + rw[n * 64 + h] * ATT_SCALE);
      ar[j] = (__bf16)(qv + rr[n * 64 + h] * ATT_SCALE);
    }
    for (int j = 0; j < 8; ++j) {
      int h = kk * 32 + half * 16 + 8 + j;
      float qv = qrow[h];
      ac[8 + j] = (__bf16)(qv + rw[n * 64 + h] * ATT_SCALE);
      ar[8 + j] = (__bf16)(qv + rr[n * 64 + h] * ATT_SCALE);
    }
    a_c[kk] = ac; a_r[kk] = ar;
  }

  // ---- token-type biases, query mask, query ids (per row) ----
  {
    const float* qp = q + ((size_t)(b * Lq + itile * 16 + ln)) * 1024 + n * 64;
    const float* sp = seg + half * 1024 + n * 64;
    float acc = 0.f;
    for (int h = 0; h < 64; ++h)
      acc += (qp[h] + rs[n * 64 + h] * ATT_SCALE) * sp[h];
    tt_lds[half][ln] = acc;
    if (half == 0) {
      float mv = 1e30f;
      for (int w = 0; w < sq; ++w)
        mv = fminf(mv, amask[b * 1024 + (itile * 16 + ln) * sq + w]);
      mr_lds[ln]  = mv;
      idq_lds[ln] = ids[b * 1024 + (itile * 16 + ln) * sq];
    }
  }
  __syncthreads();

  float ttb0[8], ttb1[8], mpen[8];
  int idqv[8];
  for (int e = 0; e < 8; ++e) {
    int m = e + 8 * half;
    ttb0[e] = tt_lds[0][m];
    ttb1[e] = tt_lds[1][m];
    mpen[e] = INF_PEN * (1.f - mr_lds[m]);
    idqv[e] = idq_lds[m];
  }

  float run_max[8], run_sum[8];
  for (int e = 0; e < 8; ++e) { run_max[e] = -1e30f; run_sum[e] = 0.f; }
  v8f outacc[4];
  outacc[0] = (v8f){}; outacc[1] = (v8f){}; outacc[2] = (v8f){}; outacc[3] = (v8f){};

  const int njt = Lk >> 4;
  for (int jt = 0; jt < njt; ++jt) {
    // ---- content scores: 2 WMMAs over DH=64 ----
    v8f cs = (v8f){};
    {
      const float* kp = k + ((size_t)(b * Lk + jt * 16 + ln)) * 1024 + n * 64;
      for (int kk = 0; kk < 2; ++kk) {
        v16bf bf;
        for (int j = 0; j < 16; ++j)
          bf[j] = (__bf16)kp[kk * 32 + half * 16 + j];
        cs = wmma_bf16(a_c[kk], bf, cs);
      }
    }
    // ---- relative-position P tile: P[i, trel] over the tile's rel window ----
    int tmin = itile * 16 - cj * (jt * 16 + 15) + Lq;
    __syncthreads();
    for (int nb = 0; nb < NPB; ++nb) {
      v8f pacc = (v8f){};
      const float* rp = Rtab + ((size_t)(tmin + nb * 16 + ln)) * 1024 + n * 64;
      for (int kk = 0; kk < 2; ++kk) {
        v16bf bf;
        for (int j = 0; j < 16; ++j)
          bf[j] = (__bf16)rp[kk * 32 + half * 16 + j];
        pacc = wmma_bf16(a_r[kk], bf, pacc);
      }
      for (int e = 0; e < 8; ++e)
        P_lds[(e + 8 * half) * 48 + nb * 16 + ln] = pacc[e];
    }
    __syncthreads();

    // ---- assemble scores (Hankel gather on P + token-type + mask) ----
    float sv[8];
    {
      int idk = ids[b * 1024 + (jt * 16 + ln) * sk];
      bool kcls = (idk == 2);
      for (int e = 0; e < 8; ++e) {
        int m = e + 8 * half;
        float posv = P_lds[m * 48 + m + cj * (15 - ln)];
        bool T = (idqv[e] == idk) || (idqv[e] == 2) || kcls;
        sv[e] = cs[e] + posv + (T ? ttb1[e] : ttb0[e]) - mpen[e];
      }
    }
    // ---- streaming softmax (row reductions across 16-lane groups) ----
    float alpha[8];
    for (int e = 0; e < 8; ++e) {
      float mx = sv[e];
      mx = fmaxf(mx, __shfl_xor(mx, 1));
      mx = fmaxf(mx, __shfl_xor(mx, 2));
      mx = fmaxf(mx, __shfl_xor(mx, 4));
      mx = fmaxf(mx, __shfl_xor(mx, 8));
      float nm = fmaxf(run_max[e], mx);
      float p  = __expf(sv[e] - nm);
      alpha[e] = __expf(run_max[e] - nm);
      run_max[e] = nm;
      float ts = p;
      ts += __shfl_xor(ts, 1);
      ts += __shfl_xor(ts, 2);
      ts += __shfl_xor(ts, 4);
      ts += __shfl_xor(ts, 8);
      run_sum[e] = run_sum[e] * alpha[e] + ts;
      sv[e] = p;
    }
    for (int t = 0; t < 4; ++t)
      for (int e = 0; e < 8; ++e)
        outacc[t][e] *= alpha[e];

    for (int e = 0; e < 8; ++e)
      p_lds[(e + 8 * half) * 16 + ln] = sv[e];
    __syncthreads();

    // ---- P @ V: zero-padded K=32 bf16 WMMA, 4 dh-blocks ----
    v16bf ap;
    for (int j = 0; j < 8; ++j)
      ap[j] = (__bf16)p_lds[ln * 16 + half * 8 + j];
    for (int j = 8; j < 16; ++j)
      ap[j] = (__bf16)0.0f;
    for (int t = 0; t < 4; ++t) {
      v16bf bv;
      for (int j = 0; j < 8; ++j)
        bv[j] = (__bf16)v[((size_t)(b * Lk + jt * 16 + half * 8 + j)) * 1024 + n * 64 + t * 16 + ln];
      for (int j = 8; j < 16; ++j) bv[j] = (__bf16)0.0f;
      outacc[t] = wmma_bf16(ap, bv, outacc[t]);
    }
    __syncthreads();
  }

  // ---- finalize ----
  for (int t = 0; t < 4; ++t) {
    for (int e = 0; e < 8; ++e) {
      int m = e + 8 * half;
      int row = itile * 16 + m;
      outv[((size_t)(b * Lq + row)) * 1024 + n * 64 + t * 16 + ln] =
          outacc[t][e] / run_sum[e];
    }
  }
}

// ===========================================================================
// Host orchestration
// ===========================================================================
struct LayerCfg { int l, Lq, Lk, sq, cj; float ss; bool cross; };

extern "C" void kernel_launch(void* const* d_in, const int* in_sizes, int n_in,
                              void* d_out, int out_size, void* d_ws, size_t ws_size,
                              hipStream_t stream)
{
  (void)in_sizes; (void)n_in; (void)out_size; (void)ws_size;

  const float* final_hidden = (const float*)d_in[0];
  const float* amask        = (const float*)d_in[4];
  const int*   ids          = (const int*)d_in[5];
  const float* Wq = (const float*)d_in[6];
  const float* bq = (const float*)d_in[7];
  const float* Wk = (const float*)d_in[8];
  const float* bk = (const float*)d_in[9];
  const float* Wv = (const float*)d_in[10];
  const float* bv = (const float*)d_in[11];
  const float* rwb = (const float*)d_in[12];
  const float* rrb = (const float*)d_in[13];
  const float* rsb = (const float*)d_in[14];
  const float* rk  = (const float*)d_in[15];
  const float* seg = (const float*)d_in[16];
  const float* Wp  = (const float*)d_in[17];
  const float* bp  = (const float*)d_in[18];
  const float* lg  = (const float*)d_in[19];
  const float* lb  = (const float*)d_in[20];

  const size_t SLOT = (size_t)BATCH * 1024 * 1024;   // floats (8 MB)
  float* ws   = (float*)d_ws;
  float* hid0 = ws;
  float* hid1 = hid0 + SLOT;
  float* up   = hid1 + SLOT;
  float* qb   = up + SLOT;
  float* kb   = qb + SLOT;
  float* vbuf = kb + SLOT;
  float* att  = vbuf + SLOT;
  float* trig = att + SLOT;                          // 2048 x 1024
  float* rtab = trig + (size_t)2048 * 1024;          // (2048+64) x 1024

  static const LayerCfg cfgs[12] = {
    {11, 256, 256, 4, 1, +1.f, false},
    {10, 256, 256, 4, 1, +1.f, false},
    { 9, 256, 256, 4, 1, +1.f, false},
    { 8, 512, 256, 2, 2, -1.f, true },
    { 7, 512, 512, 2, 1, +1.f, false},
    { 6, 512, 512, 2, 1, +1.f, false},
    { 5, 512, 512, 2, 1, +1.f, false},
    { 4,1024, 512, 1, 2, -1.f, true },
    { 3,1024,1024, 1, 1, +1.f, false},
    { 2,1024,1024, 1, 1, +1.f, false},
    { 1,1024,1024, 1, 1, +1.f, false},
    { 0,1024,1024, 1, 1, +1.f, false},
  };

  hipMemcpyAsync(hid0, final_hidden, (size_t)BATCH * 256 * 1024 * sizeof(float),
                 hipMemcpyDeviceToDevice, stream);

  float* cur = hid0;
  float* nxt = hid1;
  int Lcur = 256;

  for (int s = 0; s < 12; ++s) {
    const LayerCfg& c = cfgs[s];
    if (s == 0 || s == 4) { // block entry: upsample pre-block hidden
      size_t total = (size_t)BATCH * Lcur * 1024;
      upsample_kernel<<<(unsigned)((total + 255) / 256), 256, 0, stream>>>(cur, up, Lcur);
    }
    const float* qin = c.cross ? up : cur;
    const int l = c.l, Lq = c.Lq, Lk = c.Lk;
    const size_t wOff = (size_t)l * 1024 * 1024;
    const size_t vOff = (size_t)l * 1024;

    // projections (bf16 WMMA GEMMs)
    gemm_kernel<<<dim3(16, (BATCH * Lq) / 64), 128, 0, stream>>>(
        qin, Wq + wOff, bq + vOff, qb, BATCH * Lq, ATT_SCALE);
    gemm_kernel<<<dim3(16, (BATCH * Lk) / 64), 128, 0, stream>>>(
        cur, Wk + wOff, bk + vOff, kb, BATCH * Lk, 1.0f);
    gemm_kernel<<<dim3(16, (BATCH * Lk) / 64), 128, 0, stream>>>(
        cur, Wv + wOff, bv + vOff, vbuf, BATCH * Lk, 1.0f);

    // relative-position table: Rtab = Trig(2Lq x 1024) @ r_kernel[l]
    trig_kernel<<<dim3(4, 2 * Lq), 256, 0, stream>>>(trig, Lq, c.sq, c.ss);
    gemm_kernel<<<dim3(16, (2 * Lq) / 64), 128, 0, stream>>>(
        trig, rk + wOff, nullptr, rtab, 2 * Lq, 1.0f);

    // fused attention
    attn_kernel<<<dim3(Lq / 16, NHEAD, BATCH), 32, 0, stream>>>(
        qb, kb, vbuf, rtab, rwb + vOff, rrb + vOff, rsb + vOff,
        seg + (size_t)l * 2048, ids, amask, att, Lq, Lk, c.sq, c.sq * c.cj, c.cj);

    // post projection + residual layernorm
    gemm_kernel<<<dim3(16, (BATCH * Lq) / 64), 128, 0, stream>>>(
        att, Wp + wOff, bp + vOff, qb, BATCH * Lq, 1.0f);
    ln_kernel<<<BATCH * Lq, 256, 0, stream>>>(qb, qin, lg + vOff, lb + vOff, nxt);

    float* t = cur; cur = nxt; nxt = t;
    Lcur = Lq;
  }

  hipMemcpyAsync(d_out, cur, (size_t)BATCH * 1024 * 1024 * sizeof(float),
                 hipMemcpyDeviceToDevice, stream);
}
